// CausalSelfAttention_83614423318951
// MI455X (gfx1250) — matrix operand order repaired and checked by reading, not compiled, not run on hardware
//
#include <hip/hip_runtime.h>
#include <math.h>
#include <stdint.h>

typedef __attribute__((ext_vector_type(16))) __bf16 v16bf;
typedef __attribute__((ext_vector_type(8)))  float  v8f;
typedef __attribute__((ext_vector_type(4)))  unsigned int v4u;
typedef __attribute__((ext_vector_type(8)))  int    v8i;
typedef __attribute__((ext_vector_type(4)))  int    v4i;

#define B_   2
#define T_   2048
#define D_   2048
#define NH_  16
#define NKV_ 4
#define HD_  128
#define KV_  512   // NKV*HD

#if __has_builtin(__builtin_amdgcn_tensor_load_to_lds) && \
    __has_builtin(__builtin_amdgcn_s_wait_tensorcnt)
#define HAVE_TDM 1
#endif

union FragBF {
    v16bf v;
    uint4 q[2];
};

__device__ __forceinline__ unsigned short f2bf(float f) {
    unsigned int u = __float_as_uint(f);
    unsigned int r = u + 0x7FFFu + ((u >> 16) & 1u);  // RNE (NaN edge ignored)
    return (unsigned short)(r >> 16);
}

__device__ __forceinline__ v8f v8f_zero() {
    v8f z;
#pragma unroll
    for (int i = 0; i < 8; i++) z[i] = 0.0f;
    return z;
}

__device__ __forceinline__ v8f wmma_bf16(v16bf a, v16bf b, v8f c) {
    // D = A(16x32 bf16) * B(32x16 bf16) + C(16x16 f32)
    return __builtin_amdgcn_wmma_f32_16x16x32_bf16(
        false, a, false, b, (short)0, c, false, false);
}

#ifdef HAVE_TDM
// ---------------------------------------------------------------------------
// TDM: DMA a 2D tile (tile_d0 4B-elems wide, tile_d1 rows, row stride
// stride4 4B-elems) from global memory into LDS at byte offset lds_off.
// Tile rows are packed consecutively in LDS (tile_d0*4 bytes per row).
// Tracked by TENSORcnt. Issue from ONE wave (EXEC is ignored by TDM).
// ---------------------------------------------------------------------------
__device__ __forceinline__ void tdm_load_2d(unsigned lds_off, const void* gptr,
                                            unsigned tile_d0, unsigned tile_d1,
                                            unsigned stride4) {
    unsigned long long ga = (unsigned long long)(uintptr_t)gptr;
    v4u g0;
    g0[0] = 1u;                                            // count=1, user D#
    g0[1] = lds_off;                                       // lds_addr
    g0[2] = (unsigned)ga;                                  // global_addr[31:0]
    g0[3] = (unsigned)((ga >> 32) & 0x1FFFFFFu) | (2u << 30); // [56:32] | type=2
    v8i g1;
    g1[0] = (int)(2u << 16);              // wg_mask=0, data_size=2 (4 bytes)
    g1[1] = (int)((tile_d0 & 0xFFFFu) << 16);   // tensor_dim0[15:0] (=tile_d0)
    g1[2] = (int)((tile_d1 & 0xFFFFu) << 16);   // dim0[31:16]=0 | tensor_dim1[15:0]
    g1[3] = (int)((tile_d0 & 0xFFFFu) << 16);   // dim1[31:16]=0 | tile_dim0
    g1[4] = (int)(tile_d1 & 0xFFFFu);           // tile_dim1 | tile_dim2=0
    g1[5] = (int)stride4;                       // tensor_dim0_stride[31:0]
    g1[6] = 0;                                  // stride[47:32] | dim1_stride lo
    g1[7] = 0;
    v4i gz;
    gz[0] = gz[1] = gz[2] = gz[3] = 0;
#if __clang_major__ >= 23
    v8i gz8;
#pragma unroll
    for (int i = 0; i < 8; i++) gz8[i] = 0;
    __builtin_amdgcn_tensor_load_to_lds(g0, g1, gz, gz, gz8, 0);
#else
    __builtin_amdgcn_tensor_load_to_lds(g0, g1, gz, gz, 0);
#endif
}
#endif

// ---------------------------------------------------------------------------
// fp32 -> bf16 elementwise
// ---------------------------------------------------------------------------
__global__ void f32_to_bf16_kernel(const float* __restrict__ in,
                                   unsigned short* __restrict__ out, long n) {
    long i = (long)blockIdx.x * blockDim.x + threadIdx.x;
    long stride = (long)gridDim.x * blockDim.x;
    for (; i < n; i += stride) out[i] = f2bf(in[i]);
}

// ---------------------------------------------------------------------------
// C[M,N] (f32) = A[M,K] (bf16, row-major) @ B[N,K]^T (bf16, row-major)
// Block tile 128x128x32, 8 waves; wave = 32 rows x 64 cols = 2x4 WMMA tiles.
// A/B tiles staged to LDS by the Tensor Data Mover (double-buffered);
// falls back to cooperative global->LDS loads if the builtin is absent.
// ---------------------------------------------------------------------------
__global__ __launch_bounds__(256) void gemm_bf16_nt(
    const unsigned short* __restrict__ A,
    const unsigned short* __restrict__ Bm,
    float* __restrict__ C, int M, int N, int K) {
    __shared__ __align__(16) unsigned short sA[2][128 * 32];
    __shared__ __align__(16) unsigned short sB[2][128 * 32];

    const int tid   = threadIdx.x;
    const int lane  = tid & 31;
    const int wave  = tid >> 5;
    const int waveM = wave >> 1;   // 0..3 : 32-row strip
    const int waveN = wave & 1;    // 0..1 : 64-col strip
    const int l16   = lane & 15;
    const int lhalf = lane >> 4;

    const int bm = blockIdx.y * 128;
    const int bn = blockIdx.x * 128;

    v8f acc[2][4];
#pragma unroll
    for (int mi = 0; mi < 2; mi++)
#pragma unroll
        for (int ni = 0; ni < 4; ni++) acc[mi][ni] = v8f_zero();

    const int nk = K >> 5;  // K/32 steps

#ifdef HAVE_TDM
    const unsigned ldsA0 = (unsigned)(uintptr_t)&sA[0][0];
    const unsigned ldsA1 = (unsigned)(uintptr_t)&sA[1][0];
    const unsigned ldsB0 = (unsigned)(uintptr_t)&sB[0][0];
    const unsigned ldsB1 = (unsigned)(uintptr_t)&sB[1][0];
    if (wave == 0) {  // prologue: tiles for k-step 0 into buffer 0
        tdm_load_2d(ldsA0, A + (size_t)bm * K, 16u, 128u, (unsigned)(K >> 1));
        tdm_load_2d(ldsB0, Bm + (size_t)bn * K, 16u, 128u, (unsigned)(K >> 1));
    }
#endif

    for (int ik = 0; ik < nk; ik++) {
        const int kk  = ik << 5;
        const int cur = ik & 1;

#ifdef HAVE_TDM
        if (wave == 0) {
            if (ik + 1 < nk) {  // prefetch next tiles into the other buffer
                const unsigned la = cur ? ldsA0 : ldsA1;
                const unsigned lb = cur ? ldsB0 : ldsB1;
                tdm_load_2d(la, A + (size_t)bm * K + kk + 32, 16u, 128u,
                            (unsigned)(K >> 1));
                tdm_load_2d(lb, Bm + (size_t)bn * K + kk + 32, 16u, 128u,
                            (unsigned)(K >> 1));
                // TDM completes in order: <=2 outstanding => current tiles done
                __builtin_amdgcn_s_wait_tensorcnt(2);
            } else {
                __builtin_amdgcn_s_wait_tensorcnt(0);
            }
        }
        __syncthreads();
#else
        // cooperative staging fallback
#pragma unroll
        for (int u = 0; u < 2; u++) {
            int unit = tid + u * 256;          // 0..511
            int row  = unit >> 2;              // 0..127
            int col8 = (unit & 3) * 8;         // bf16 elems (16B units)
            *(uint4*)&sA[cur][row * 32 + col8] =
                *(const uint4*)&A[(size_t)(bm + row) * K + kk + col8];
            *(uint4*)&sB[cur][row * 32 + col8] =
                *(const uint4*)&Bm[(size_t)(bn + row) * K + kk + col8];
        }
        __syncthreads();
#endif

        FragBF af[2], bfr[4];
#pragma unroll
        for (int mi = 0; mi < 2; mi++) {
            int row = waveM * 32 + mi * 16 + l16;
            int klo = lhalf ? 8 : 0;  // A layout: K {klo..klo+7, klo+16..klo+23}
            af[mi].q[0] = *(const uint4*)&sA[cur][row * 32 + klo];
            af[mi].q[1] = *(const uint4*)&sA[cur][row * 32 + klo + 16];
        }
#pragma unroll
        for (int ni = 0; ni < 4; ni++) {
            int col = waveN * 64 + ni * 16 + l16;
            int klo = lhalf ? 8 : 0;  // B layout: K {klo..klo+7, klo+16..klo+23}
            bfr[ni].q[0] = *(const uint4*)&sB[cur][col * 32 + klo];
            bfr[ni].q[1] = *(const uint4*)&sB[cur][col * 32 + klo + 16];
        }
#pragma unroll
        for (int mi = 0; mi < 2; mi++)
#pragma unroll
            for (int ni = 0; ni < 4; ni++)
                acc[mi][ni] = wmma_bf16(af[mi].v, bfr[ni].v, acc[mi][ni]);
        __syncthreads();
    }

#pragma unroll
    for (int mi = 0; mi < 2; mi++)
#pragma unroll
        for (int ni = 0; ni < 4; ni++)
#pragma unroll
            for (int r = 0; r < 8; r++) {
                int row = bm + waveM * 32 + mi * 16 + r + lhalf * 8;
                int col = bn + waveN * 64 + ni * 16 + l16;
                C[(size_t)row * N + col] = acc[mi][ni][r];
            }
}

// ---------------------------------------------------------------------------
// RMSNorm + partial RoPE + gain (q) / RMSNorm + RoPE (k) / v += ve (v)
// One wave per head-task; 4 fp32 values per lane (HD=128).
// Emits bf16: Q [B,NH,T,HD], K [B,NKV,T,HD], V transposed [B,NKV,HD,T].
// ---------------------------------------------------------------------------
__global__ __launch_bounds__(256) void qkv_post(
    const float* __restrict__ qf, const float* __restrict__ kf,
    const float* __restrict__ vf, const float* __restrict__ ve,
    const float* __restrict__ gain,
    unsigned short* __restrict__ Qb, unsigned short* __restrict__ Kb,
    unsigned short* __restrict__ Vt) {
    const int bt   = blockIdx.x;       // 0..B*T-1
    const int b    = bt / T_;
    const int t    = bt % T_;
    const int wave = threadIdx.x >> 5;
    const int lane = threadIdx.x & 31;

    for (int task = wave; task < NH_ + 2 * NKV_; task += 8) {
        if (task < NH_ + NKV_) {
            // ---- q or k: norm + rope ----
            const bool isq = task < NH_;
            const int  h   = isq ? task : task - NH_;
            const float* src = isq ? (qf + (size_t)bt * D_ + h * HD_)
                                   : (kf + (size_t)bt * KV_ + h * HD_);
            float x[4];
            float ssq = 0.0f;
#pragma unroll
            for (int i = 0; i < 4; i++) {
                x[i] = src[lane * 4 + i];
                ssq += x[i] * x[i];
            }
            ssq += __shfl_xor(ssq, 1, 32);
            ssq += __shfl_xor(ssq, 2, 32);
            ssq += __shfl_xor(ssq, 4, 32);
            ssq += __shfl_xor(ssq, 8, 32);
            ssq += __shfl_xor(ssq, 16, 32);
            const float nrm = rsqrtf(ssq * (1.0f / HD_) + 1.1920929e-7f);
#pragma unroll
            for (int i = 0; i < 4; i++) x[i] *= nrm;
            // partial rope on dims [0,64): pair (j, j+32) lives in lanes L, L^8
#pragma unroll
            for (int i = 0; i < 4; i++) {
                int d = lane * 4 + i;
                float other = __shfl_xor(x[i], 8, 32);
                if (d < 64) {
                    int j = d & 31;
                    float f = (float)t * __powf(10000.0f, -(float)j * (1.0f / 64.0f));
                    float c, s;
                    __sincosf(f, &s, &c);
                    x[i] = (d < 32) ? (x[i] * c - other * s)
                                    : (other * s + x[i] * c);
                }
            }
            if (isq) {
                const float g = gain[h];
                unsigned short* dst = Qb + ((size_t)(b * NH_ + h) * T_ + t) * HD_;
#pragma unroll
                for (int i = 0; i < 4; i++) dst[lane * 4 + i] = f2bf(x[i] * g);
            } else {
                unsigned short* dst = Kb + ((size_t)(b * NKV_ + h) * T_ + t) * HD_;
#pragma unroll
                for (int i = 0; i < 4; i++) dst[lane * 4 + i] = f2bf(x[i]);
            }
        } else {
            // ---- v: add ve, store transposed [B,NKV,HD,T] ----
            const int h = task - NH_ - NKV_;
#pragma unroll
            for (int i = 0; i < 4; i++) {
                int d = lane * 4 + i;
                size_t s = (size_t)bt * KV_ + h * HD_ + d;
                float val = vf[s] + ve[s];
                Vt[((size_t)(b * NKV_ + h) * HD_ + d) * T_ + t] = f2bf(val);
            }
        }
    }
}

// ---------------------------------------------------------------------------
// Causal GQA flash attention. One wave = one 16-row Q tile, 32 keys/step.
// Q [B,NH,T,HD] bf16; K [B,NKV,T,HD] bf16; V^T [B,NKV,HD,T] bf16.
// Y [B,T,NH*HD] bf16 (feeds proj GEMM directly).
// ---------------------------------------------------------------------------
__global__ __launch_bounds__(256) void flash_attn(
    const unsigned short* __restrict__ Q,
    const unsigned short* __restrict__ Kt,
    const unsigned short* __restrict__ Vt,
    unsigned short* __restrict__ Y) {
    __shared__ __align__(16) unsigned short sP[8][16 * 32];  // per-wave P staging

    const int lane  = threadIdx.x & 31;
    const int wave  = threadIdx.x >> 5;
    const int l16   = lane & 15;
    const int lhalf = lane >> 4;
    const int h     = blockIdx.y;
    const int b     = blockIdx.z;
    const int kv    = h / (NH_ / NKV_);
    const int r0    = (blockIdx.x * 8 + wave) * 16;

    const size_t qbase = (size_t)(b * NH_ + h) * T_ * HD_;
    const size_t kbase = (size_t)(b * NKV_ + kv) * T_ * HD_;
    const size_t vbase = (size_t)(b * NKV_ + kv) * HD_ * T_;

    // Q A-fragments: 4 chunks of K=32 over HD=128
    FragBF qfr[4];
    {
        const unsigned short* qrow = Q + qbase + (size_t)(r0 + l16) * HD_;
#pragma unroll
        for (int kc = 0; kc < 4; kc++) {
            int klo = kc * 32 + (lhalf ? 8 : 0);
            qfr[kc].q[0] = *(const uint4*)&qrow[klo];
            qfr[kc].q[1] = *(const uint4*)&qrow[klo + 16];
        }
    }

    v8f acc[8];
#pragma unroll
    for (int ni = 0; ni < 8; ni++) acc[ni] = v8f_zero();
    float mrow[8], lrow[8];
#pragma unroll
    for (int r = 0; r < 8; r++) { mrow[r] = -INFINITY; lrow[r] = 0.0f; }

    const float scale = 0.08838834764831845f;  // 1/sqrt(128)
    const int ng = (r0 + 16 + 31) / 32;        // 32-key groups covering keys <= r0+15

    for (int g = 0; g < ng; g++) {
        const int c0 = g * 32;
        v8f s[2];
#pragma unroll
        for (int hh = 0; hh < 2; hh++) {
            const int cb = c0 + hh * 16;
            if (cb > r0 + 15) {
#pragma unroll
                for (int r = 0; r < 8; r++) s[hh][r] = -INFINITY;
                continue;
            }
            v8f sv = v8f_zero();
            const unsigned short* krow = Kt + kbase + (size_t)(cb + l16) * HD_;
#pragma unroll
            for (int kc = 0; kc < 4; kc++) {
                FragBF kfr;
                int klo = kc * 32 + (lhalf ? 8 : 0);
                kfr.q[0] = *(const uint4*)&krow[klo];
                kfr.q[1] = *(const uint4*)&krow[klo + 16];
                sv = wmma_bf16(qfr[kc].v, kfr.v, sv);
            }
#pragma unroll
            for (int r = 0; r < 8; r++) {
                float v   = sv[r] * scale;
                int   row = r0 + r + lhalf * 8;
                int   key = cb + l16;
                s[hh][r]  = (key <= row) ? v : -INFINITY;
            }
        }

        // online softmax stats over the 32-key group
        float mnew[8], alpha[8];
#pragma unroll
        for (int r = 0; r < 8; r++) {
            float v = fmaxf(s[0][r], s[1][r]);
            v = fmaxf(v, __shfl_xor(v, 1, 32));
            v = fmaxf(v, __shfl_xor(v, 2, 32));
            v = fmaxf(v, __shfl_xor(v, 4, 32));
            v = fmaxf(v, __shfl_xor(v, 8, 32));
            mnew[r]  = fmaxf(mrow[r], v);
            alpha[r] = __expf(mrow[r] - mnew[r]);
            mrow[r]  = mnew[r];
        }

        // P = exp(S - m) -> bf16 into LDS (C-layout -> row-major 16x32)
        unsigned short* pw = &sP[wave][0];
        float psum[8];
#pragma unroll
        for (int r = 0; r < 8; r++) psum[r] = 0.0f;
#pragma unroll
        for (int hh = 0; hh < 2; hh++)
#pragma unroll
            for (int r = 0; r < 8; r++) {
                float p = __expf(s[hh][r] - mnew[r]);
                psum[r] += p;
                pw[(r + lhalf * 8) * 32 + hh * 16 + l16] = f2bf(p);
            }
#pragma unroll
        for (int r = 0; r < 8; r++) {
            float t = psum[r];
            t += __shfl_xor(t, 1, 32);
            t += __shfl_xor(t, 2, 32);
            t += __shfl_xor(t, 4, 32);
            t += __shfl_xor(t, 8, 32);
            lrow[r] = lrow[r] * alpha[r] + t;
        }
#pragma unroll
        for (int ni = 0; ni < 8; ni++)
#pragma unroll
            for (int r = 0; r < 8; r++) acc[ni][r] *= alpha[r];

        // in-wave LDS RAW: stores above -> A-fragment reads below
        asm volatile("s_wait_dscnt 0x0" ::: "memory");

        FragBF pfr;
        {
            int row = l16;
            int klo = lhalf ? 8 : 0;
            pfr.q[0] = *(const uint4*)&pw[row * 32 + klo];
            pfr.q[1] = *(const uint4*)&pw[row * 32 + klo + 16];
        }
        // Y(16x128) += P(16x32) @ V(32x128): V^T gives contiguous-key B-frags
#pragma unroll
        for (int ni = 0; ni < 8; ni++) {
            FragBF vfr;
            const unsigned short* vrow =
                Vt + vbase + (size_t)(ni * 16 + l16) * T_ + c0 + (lhalf ? 8 : 0);
            vfr.q[0] = *(const uint4*)&vrow[0];
            vfr.q[1] = *(const uint4*)&vrow[16];
            acc[ni] = wmma_bf16(pfr.v, vfr.v, acc[ni]);
        }
    }

    // normalize and write bf16 y at [b, t, h*HD + d]
    float inv[8];
#pragma unroll
    for (int r = 0; r < 8; r++) inv[r] = 1.0f / lrow[r];
#pragma unroll
    for (int ni = 0; ni < 8; ni++)
#pragma unroll
        for (int r = 0; r < 8; r++) {
            int row = r0 + r + lhalf * 8;
            int col = h * HD_ + ni * 16 + l16;
            Y[(size_t)(b * T_ + row) * D_ + col] = f2bf(acc[ni][r] * inv[r]);
        }
}

// ---------------------------------------------------------------------------
extern "C" void kernel_launch(void* const* d_in, const int* in_sizes, int n_in,
                              void* d_out, int out_size, void* d_ws, size_t ws_size,
                              hipStream_t stream) {
    const float* x     = (const float*)d_in[0];
    const float* ve    = (const float*)d_in[1];
    const float* Wq    = (const float*)d_in[2];
    const float* Wk    = (const float*)d_in[3];
    const float* Wv    = (const float*)d_in[4];
    const float* Wp    = (const float*)d_in[5];
    const float* qgain = (const float*)d_in[6];
    float* out = (float*)d_out;

    const int M = B_ * T_;  // 4096

    // bump allocator over workspace (~130 MB total)
    char* ws = (char*)d_ws;
    auto alloc = [&](size_t bytes) -> char* {
        char* p = ws;
        ws += (bytes + 255) & ~(size_t)255;
        return p;
    };
    unsigned short* x_bf  = (unsigned short*)alloc((size_t)M * D_ * 2);
    unsigned short* wq_bf = (unsigned short*)alloc((size_t)D_ * D_ * 2);
    unsigned short* wk_bf = (unsigned short*)alloc((size_t)KV_ * D_ * 2);
    unsigned short* wv_bf = (unsigned short*)alloc((size_t)KV_ * D_ * 2);
    unsigned short* wp_bf = (unsigned short*)alloc((size_t)D_ * D_ * 2);
    float* q_f = (float*)alloc((size_t)M * D_ * 4);
    float* k_f = (float*)alloc((size_t)M * KV_ * 4);
    float* v_f = (float*)alloc((size_t)M * KV_ * 4);
    unsigned short* q_bf  = (unsigned short*)alloc((size_t)M * D_ * 2);
    unsigned short* k_bf  = (unsigned short*)alloc((size_t)M * KV_ * 2);
    unsigned short* vT_bf = (unsigned short*)alloc((size_t)M * KV_ * 2);
    unsigned short* y_bf  = (unsigned short*)alloc((size_t)M * D_ * 2);

    // 1) convert inputs to bf16
    f32_to_bf16_kernel<<<2048, 256, 0, stream>>>(x, x_bf, (long)M * D_);
    f32_to_bf16_kernel<<<2048, 256, 0, stream>>>(Wq, wq_bf, (long)D_ * D_);
    f32_to_bf16_kernel<<<512, 256, 0, stream>>>(Wk, wk_bf, (long)KV_ * D_);
    f32_to_bf16_kernel<<<512, 256, 0, stream>>>(Wv, wv_bf, (long)KV_ * D_);
    f32_to_bf16_kernel<<<2048, 256, 0, stream>>>(Wp, wp_bf, (long)D_ * D_);

    // 2) QKV projection GEMMs (C = A @ W^T)
    {
        dim3 gq(D_ / 128, M / 128);
        gemm_bf16_nt<<<gq, 256, 0, stream>>>(x_bf, wq_bf, q_f, M, D_, D_);
        dim3 gk(KV_ / 128, M / 128);
        gemm_bf16_nt<<<gk, 256, 0, stream>>>(x_bf, wk_bf, k_f, M, KV_, D_);
        gemm_bf16_nt<<<gk, 256, 0, stream>>>(x_bf, wv_bf, v_f, M, KV_, D_);
    }

    // 3) RMSNorm + RoPE + gain, v += ve; emit attention-friendly bf16 layouts
    qkv_post<<<M, 256, 0, stream>>>(q_f, k_f, v_f, ve, qgain, q_bf, k_bf, vT_bf);

    // 4) causal GQA flash attention
    {
        dim3 ga(T_ / 128, NH_, B_);
        flash_attn<<<ga, 256, 0, stream>>>(q_bf, k_bf, vT_bf, y_bf);
    }

    // 5) output projection -> fp32 d_out
    {
        dim3 gp(D_ / 128, M / 128);
        gemm_bf16_nt<<<gp, 256, 0, stream>>>(y_bf, wp_bf, out, M, D_, D_);
    }
}
